// GraphNeuralNetwork_50251117363558
// MI455X (gfx1250) — matrix operand order repaired and checked by reading, not compiled, not run on hardware
//
#include <hip/hip_runtime.h>

// GNN message passing, rewritten via linearity of segment_sum:
//   agg[v] = S[v] @ Wtop + deg[v] * (h[v] @ Wbot + b_msg),  S[v] = sum_{dst(e)=v} h[src[e]]
// so each hop = L2-resident scatter-add + two [N,512]@[512,256] WMMA GEMMs.
// GEMM: 64x64 block tile, 4 waves, f32->f16 staged in double-buffered LDS,
// v_wmma_f32_16x16x32_f16 with f32 accumulate.

typedef __attribute__((ext_vector_type(16))) _Float16 v16h;
typedef __attribute__((ext_vector_type(8)))  _Float16 v8h;
typedef __attribute__((ext_vector_type(4)))  _Float16 v4h;
typedef __attribute__((ext_vector_type(8)))  float    v8f;

#define GNN_D     256
#define TILE_M    64
#define TILE_N    64
#define LDA       40   // halves per LDS row (32 data + 8 pad); 80B = 20 dwords

__device__ __forceinline__ void atomic_add_f32(float* p, float v) {
    __hip_atomic_fetch_add(p, v, __ATOMIC_RELAXED, __HIP_MEMORY_SCOPE_AGENT);
}

// ---- GEMM tile staging helpers (global f32 -> regs -> LDS f16) -------------

__device__ __forceinline__ void load_a_regs(const float* __restrict__ A0,
                                            const float* __restrict__ A1,
                                            const float* __restrict__ rowScale,
                                            int N, int row0, int t, int ks,
                                            float4 (&ar)[4])
{
    const float* Asrc = (ks < 8) ? A0 : A1;
    const int kk0 = (ks & 7) * 32;
    const bool doScale = (ks >= 8) && (rowScale != nullptr);
    #pragma unroll
    for (int j = 0; j < 4; ++j) {
        int q  = t + j * 128;           // 0..511
        int r  = q >> 3;                // tile row 0..63
        int c4 = (q & 7) << 2;          // col 0,4,...,28
        int gr = row0 + r;
        if (gr >= N) gr = N - 1;        // clamp: OOB rows are never stored, so
                                        // duplicated data is harmless (branch-free)
        float4 v = *reinterpret_cast<const float4*>(Asrc + (size_t)gr * GNN_D + kk0 + c4);
        if (doScale) {
            float s = rowScale[gr];
            v.x *= s; v.y *= s; v.z *= s; v.w *= s;
        }
        ar[j] = v;
    }
}

__device__ __forceinline__ void store_a_lds(_Float16* __restrict__ sA, int t,
                                            const float4 (&ar)[4])
{
    #pragma unroll
    for (int j = 0; j < 4; ++j) {
        int q  = t + j * 128;
        int r  = q >> 3;
        int c4 = (q & 7) << 2;
        v4h p;
        p[0] = (_Float16)ar[j].x; p[1] = (_Float16)ar[j].y;
        p[2] = (_Float16)ar[j].z; p[3] = (_Float16)ar[j].w;
        *reinterpret_cast<v4h*>(&sA[r * LDA + c4]) = p;   // 8B aligned ds_store_b64
    }
}

__device__ __forceinline__ void load_b_regs(const float* __restrict__ B,
                                            int c0, int t, int ks,
                                            float4 (&br)[4])
{
    const int k0 = ks * 32;
    #pragma unroll
    for (int j = 0; j < 4; ++j) {
        int q  = t + j * 128;           // 0..511
        int k  = q >> 4;                // 0..31
        int n4 = (q & 15) << 2;         // 0,4,...,60
        br[j] = *reinterpret_cast<const float4*>(B + (size_t)(k0 + k) * GNN_D + c0 + n4);
    }
}

__device__ __forceinline__ void store_b_lds(_Float16* __restrict__ sB, int t,
                                            const float4 (&br)[4])
{
    #pragma unroll
    for (int j = 0; j < 4; ++j) {
        int q  = t + j * 128;
        int k  = q >> 4;
        int n4 = (q & 15) << 2;
        sB[(n4 + 0) * LDA + k] = (_Float16)br[j].x;   // transposed: sB[n][k]
        sB[(n4 + 1) * LDA + k] = (_Float16)br[j].y;
        sB[(n4 + 2) * LDA + k] = (_Float16)br[j].z;
        sB[(n4 + 3) * LDA + k] = (_Float16)br[j].w;
    }
}

// ---------------------------------------------------------------------------
// Unified GEMM: Out[m][n] = post( A0[m,:]@B[0:256,:] + s(m)*A1[m,:]@B[256:512,:]
//                                 + bs(m)*bias[n] )
//   s(m)  = rowScale[m] if rowScale else 1
//   bs(m) = rowScale[m] if scaleBias else 1
//   post  = relu if relu else identity
// ---------------------------------------------------------------------------
__global__ __launch_bounds__(128)
void gnn_gemm_kernel(const float* __restrict__ A0,
                     const float* __restrict__ A1,
                     const float* __restrict__ rowScale,
                     const float* __restrict__ B,      // [512, 256] row-major (K-major)
                     const float* __restrict__ bias,   // [256]
                     float* __restrict__ Out,          // [N, 256]
                     int N, int scaleBias, int relu)
{
    __shared__ __align__(16) _Float16 sA[2][TILE_M * LDA];   // double-buffered
    __shared__ __align__(16) _Float16 sB[2][TILE_N * LDA];

    const int t    = threadIdx.x;
    const int wave = t >> 5;
    const int lane = t & 31;
    const int row0 = blockIdx.x * TILE_M;
    const int c0   = blockIdx.y * TILE_N;

    v8f acc[4] = {};

    // prologue: stage K-step 0
    {
        float4 ar[4], br[4];
        load_a_regs(A0, A1, rowScale, N, row0, t, 0, ar);
        load_b_regs(B, c0, t, 0, br);
        store_a_lds(sA[0], t, ar);
        store_b_lds(sB[0], t, br);
    }

    // fragment addressing (contiguous K order, same for A and B):
    //   A: lane holds row lane%16, contiguous halves K = ab+0..15,    ab = lane<16?0:16
    //   B: lane holds col lane%16, contiguous halves K = kb+0..15,    kb = lane<16?0:16
    const int ab = (lane < 16) ? 0 : 16;
    const int kb = (lane < 16) ? 0 : 16;
    const int am = wave * 16 + (lane & 15);

    for (int ks = 0; ks < 16; ++ks) {
        __syncthreads();
        const int cur = ks & 1;
        float4 nar[4], nbr[4];
        if (ks < 15) {                              // issue next tile's global loads
            load_a_regs(A0, A1, rowScale, N, row0, t, ks + 1, nar);
            load_b_regs(B, c0, t, ks + 1, nbr);
        }

        const _Float16* cA = sA[cur];
        const _Float16* cB = sB[cur];
        v8h alo = *reinterpret_cast<const v8h*>(&cA[am * LDA + ab]);
        v8h ahi = *reinterpret_cast<const v8h*>(&cA[am * LDA + ab + 8]);
        v16h afrag = __builtin_shufflevector(alo, ahi,
                        0,1,2,3,4,5,6,7,8,9,10,11,12,13,14,15);
        #pragma unroll
        for (int tn = 0; tn < 4; ++tn) {
            int bn = tn * 16 + (lane & 15);
            v8h blo = *reinterpret_cast<const v8h*>(&cB[bn * LDA + kb]);
            v8h bhi = *reinterpret_cast<const v8h*>(&cB[bn * LDA + kb + 8]);
            v16h bfrag = __builtin_shufflevector(blo, bhi,
                            0,1,2,3,4,5,6,7,8,9,10,11,12,13,14,15);
            acc[tn] = __builtin_amdgcn_wmma_f32_16x16x32_f16(
                false, afrag, false, bfrag, (short)0, acc[tn], false, false);
        }

        if (ks < 15) {                              // fill the other buffer
            store_a_lds(sA[cur ^ 1], t, nar);
            store_b_lds(sB[cur ^ 1], t, nbr);
        }
    }

    // epilogue: C layout VGPR r -> M = r + (lane<16?0:8), N = lane%16
    const int mbase = row0 + wave * 16 + ((lane >> 4) << 3);
    const int nloc  = lane & 15;
    #pragma unroll
    for (int tn = 0; tn < 4; ++tn) {
        int n = c0 + tn * 16 + nloc;
        float bv = bias[n];
        #pragma unroll
        for (int r = 0; r < 8; ++r) {
            int m = mbase + r;
            if (m < N) {
                float bs  = scaleBias ? rowScale[m] : 1.0f;
                float val = acc[tn][r] + bs * bv;
                if (relu) val = fmaxf(val, 0.0f);
                Out[(size_t)m * GNN_D + n] = val;
            }
        }
    }
}

// h0 = nodes @ W_proj + b_proj + type_emb[types]
__global__ __launch_bounds__(256)
void gnn_init_kernel(const float* __restrict__ nodes, const int* __restrict__ types,
                     const float* __restrict__ type_emb, const float* __restrict__ W_proj,
                     const float* __restrict__ b_proj, float* __restrict__ h, int N)
{
    __shared__ float nd[32];
    int m = blockIdx.x;
    int n = threadIdx.x;
    if (threadIdx.x < 32) nd[threadIdx.x] = nodes[(size_t)m * 32 + threadIdx.x];
    __syncthreads();
    float acc = b_proj[n] + type_emb[(size_t)types[m] * GNN_D + n];
    #pragma unroll
    for (int k = 0; k < 32; ++k) acc += nd[k] * W_proj[k * GNN_D + n];
    h[(size_t)m * GNN_D + n] = acc;
}

__global__ __launch_bounds__(256)
void gnn_zero_kernel(float* __restrict__ p, int n)
{
    int i = blockIdx.x * 256 + threadIdx.x;
    if (i < n) p[i] = 0.0f;
}

__global__ __launch_bounds__(256)
void gnn_deg_kernel(const int* __restrict__ dst, float* __restrict__ deg, int E)
{
    int e = blockIdx.x * 256 + threadIdx.x;
    if (e < E) atomic_add_f32(&deg[dst[e]], 1.0f);
}

// S[dst[e]] += h[src[e]]   — one wave per edge, 8 floats per lane
__global__ __launch_bounds__(256)
void gnn_scatter_kernel(const float* __restrict__ h, const int* __restrict__ src,
                        const int* __restrict__ dst, float* __restrict__ S, int E)
{
    int e = blockIdx.x * 8 + (threadIdx.x >> 5);
    if (e >= E) return;
    int lane = threadIdx.x & 31;
    int s = src[e], d = dst[e];
    const float4* hp = reinterpret_cast<const float4*>(h + (size_t)s * GNN_D) + lane * 2;
    float4 x0 = hp[0], x1 = hp[1];
    float* Sp = S + (size_t)d * GNN_D + lane * 8;
    atomic_add_f32(Sp + 0, x0.x); atomic_add_f32(Sp + 1, x0.y);
    atomic_add_f32(Sp + 2, x0.z); atomic_add_f32(Sp + 3, x0.w);
    atomic_add_f32(Sp + 4, x1.x); atomic_add_f32(Sp + 5, x1.y);
    atomic_add_f32(Sp + 6, x1.z); atomic_add_f32(Sp + 7, x1.w);
}

// emb[c] = max over nodes of h[:, c]  (one block per column)
__global__ __launch_bounds__(256)
void gnn_max_kernel(const float* __restrict__ h, float* __restrict__ emb, int N)
{
    __shared__ float red[256];
    int c = blockIdx.x;
    float m = -3.4e38f;
    for (int r = threadIdx.x; r < N; r += 256)
        m = fmaxf(m, h[(size_t)r * GNN_D + c]);
    red[threadIdx.x] = m;
    __syncthreads();
    for (int s = 128; s > 0; s >>= 1) {
        if (threadIdx.x < s)
            red[threadIdx.x] = fmaxf(red[threadIdx.x], red[threadIdx.x + s]);
        __syncthreads();
    }
    if (threadIdx.x == 0) emb[c] = red[0];
}

// out = emb @ W_out + b_out
__global__ __launch_bounds__(256)
void gnn_out_kernel(const float* __restrict__ emb, const float* __restrict__ W_out,
                    const float* __restrict__ b_out, float* __restrict__ out)
{
    __shared__ float e[256];
    e[threadIdx.x] = emb[threadIdx.x];
    __syncthreads();
    float acc = b_out[threadIdx.x];
    for (int k = 0; k < 256; ++k) acc += e[k] * W_out[k * GNN_D + threadIdx.x];
    out[threadIdx.x] = acc;
}

extern "C" void kernel_launch(void* const* d_in, const int* in_sizes, int n_in,
                              void* d_out, int out_size, void* d_ws, size_t ws_size,
                              hipStream_t stream)
{
    const float* nodes    = (const float*)d_in[0];
    const int*   edges    = (const int*)d_in[1];
    const int*   types    = (const int*)d_in[2];
    const float* type_emb = (const float*)d_in[3];
    const float* W_proj   = (const float*)d_in[4];
    const float* b_proj   = (const float*)d_in[5];
    const float* W_msg    = (const float*)d_in[6];
    const float* b_msg    = (const float*)d_in[7];
    const float* W_upd    = (const float*)d_in[8];
    const float* b_upd    = (const float*)d_in[9];
    const float* W_out    = (const float*)d_in[10];
    const float* b_out    = (const float*)d_in[11];
    float* out = (float*)d_out;

    const int N = in_sizes[2];       // 25000 nodes (node_types count)
    const int E = in_sizes[1] / 2;   // 400000 edges
    const int* src = edges;
    const int* dst = edges + E;

    // workspace: ping-pong h/S buffers + agg + deg + emb  (~77 MB)
    float* buf0 = (float*)d_ws;
    float* buf1 = buf0 + (size_t)N * GNN_D;
    float* agg  = buf1 + (size_t)N * GNN_D;
    float* deg  = agg  + (size_t)N * GNN_D;
    float* emb  = deg  + N;

    const int nd = N * GNN_D;
    dim3 gemmGrid((N + TILE_M - 1) / TILE_M, GNN_D / TILE_N);

    // degrees (dst is fixed across hops)
    gnn_zero_kernel<<<(N + 255) / 256, 256, 0, stream>>>(deg, N);
    gnn_deg_kernel<<<(E + 255) / 256, 256, 0, stream>>>(dst, deg, E);

    // h0
    gnn_init_kernel<<<N, 256, 0, stream>>>(nodes, types, type_emb, W_proj, b_proj, buf0, N);

    float* h  = buf0;
    float* hn = buf1;   // doubles as S (scatter sum) then as h_{i+1}
    for (int i = 0; i < 4; ++i) {
        gnn_zero_kernel<<<(nd + 255) / 256, 256, 0, stream>>>(hn, nd);
        gnn_scatter_kernel<<<(E + 7) / 8, 256, 0, stream>>>(h, src, dst, hn, E);
        // agg = S@Wtop + (deg.*h)@Wbot + deg.*b_msg
        gnn_gemm_kernel<<<gemmGrid, 128, 0, stream>>>(
            hn, h, deg, W_msg + (size_t)i * 512 * GNN_D, b_msg + i * GNN_D,
            agg, N, /*scaleBias=*/1, /*relu=*/0);
        // h_new = relu([h, agg]@W_upd + b_upd)   (written over S)
        gnn_gemm_kernel<<<gemmGrid, 128, 0, stream>>>(
            h, agg, nullptr, W_upd + (size_t)i * 512 * GNN_D, b_upd + i * GNN_D,
            hn, N, /*scaleBias=*/0, /*relu=*/1);
        float* tmp = h; h = hn; hn = tmp;
    }

    gnn_max_kernel<<<GNN_D, 256, 0, stream>>>(h, emb, N);
    gnn_out_kernel<<<1, 256, 0, stream>>>(emb, W_out, b_out, out);
}
